// AttentionLayer_88399016887055
// MI455X (gfx1250) — compile-verified, matrix operand order checked by reading, not run on hardware
//
#include <hip/hip_runtime.h>
#include <hip/hip_bf16.h>

#define FT 512
#define NN 4096
#define HH 4
#define KT 64            // K-halfs staged per LDS buffer
#define LDA 72           // 64 + 8 pad halfs -> 144B rows (16B aligned)

typedef __attribute__((ext_vector_type(16))) _Float16 v16h;
typedef __attribute__((ext_vector_type(8)))  _Float16 v8h;
typedef __attribute__((ext_vector_type(8)))  float    v8f;

union F16Frag { v16h v; v8h v8[2]; };
union F32Acc  { v8f v; float f[8]; };

// ---- CDNA5 async memory->LDS copy (ASYNCcnt-tracked, no VGPR staging) ----
__device__ __forceinline__ void async_copy_b128(unsigned lds_addr, const void* gaddr) {
  asm volatile("global_load_async_to_lds_b128 %0, %1, off"
               :: "v"(lds_addr), "v"(gaddr)
               : "memory");
}
__device__ __forceinline__ void wait_async0() {
  asm volatile("s_wait_asynccnt 0x0" ::: "memory");
}
__device__ __forceinline__ unsigned lds_off(const void* p) {
  // generic LDS pointer: low 32 bits are the LDS byte offset (aperture mapping)
  return (unsigned)(unsigned long long)p;
}

// ---------------- Kernel 1: per-head 1x1 conv (projection) ----------------
__global__ void gat_proj_kernel(const float* __restrict__ ft,
                                const float* __restrict__ W,
                                const float* __restrict__ bvec,
                                float* __restrict__ w) {
  const int n = blockIdx.x * blockDim.x + threadIdx.x;
  const int h = blockIdx.y;
  float acc = 0.f;
#pragma unroll 8
  for (int f = 0; f < FT; ++f)
    acc = fmaf(W[h * FT + f], ft[(size_t)f * NN + n], acc);
  w[h * NN + n] = acc + bvec[h];
}

// ---------------- Kernel 2: fused lrelu(w_i+w_j)+bias softmax -> f16 ------
__global__ __launch_bounds__(256) void gat_softmax_kernel(
    const float* __restrict__ bias_mat,
    const float* __restrict__ w,
    _Float16* __restrict__ a) {
  const int row = blockIdx.x;        // 0 .. H*N-1
  const int h   = row >> 12;
  const int i   = row & (NN - 1);
  const int tid = threadIdx.x;
  const int j0  = tid * 16;

  const float wi = w[h * NN + i];
  const float* __restrict__ brow = bias_mat + (size_t)i * NN;
  const float* __restrict__ wrow = w + h * NN;

  float sval[16];
  float lmax = -__builtin_inff();
#pragma unroll
  for (int k = 0; k < 16; ++k) {
    const int j = j0 + k;
    float s = wi + wrow[j];
    s = s > 0.f ? s : 0.2f * s;      // leaky_relu(0.2)
    s += brow[j];
    sval[k] = s;
    lmax = fmaxf(lmax, s);
  }

  __shared__ float red[256];
  red[tid] = lmax;
  __syncthreads();
  for (int s = 128; s > 0; s >>= 1) {
    if (tid < s) red[tid] = fmaxf(red[tid], red[tid + s]);
    __syncthreads();
  }
  const float rmax = red[0];
  __syncthreads();

  float lsum = 0.f;
#pragma unroll
  for (int k = 0; k < 16; ++k) {
    sval[k] = __expf(sval[k] - rmax);
    lsum += sval[k];
  }
  red[tid] = lsum;
  __syncthreads();
  for (int s = 128; s > 0; s >>= 1) {
    if (tid < s) red[tid] += red[tid + s];
    __syncthreads();
  }
  const float inv = 1.0f / red[0];

  _Float16* __restrict__ arow = a + (size_t)row * NN;
#pragma unroll
  for (int k = 0; k < 16; ++k)
    arow[j0 + k] = (_Float16)(sval[k] * inv);
}

// ------- Kernel 3: t[h] = a[h]*a[h]^T, WMMA f16, K=64 async double buffer -
__global__ __launch_bounds__(256) void gat_syrk_kernel(
    const _Float16* __restrict__ a,
    float* __restrict__ out) {
  const int h   = blockIdx.z;
  const int im0 = blockIdx.y * 128;   // output rows
  const int in0 = blockIdx.x * 128;   // output cols (also rows of a)
  const _Float16* __restrict__ ah = a + (size_t)h * NN * NN;

  __shared__ __align__(16) _Float16 As[2][128][LDA];  // 2 x 18 KB
  __shared__ __align__(16) _Float16 Bs[2][128][LDA];  // 2 x 18 KB

  const int tid  = threadIdx.x;
  const int lane = tid & 31;
  const int wv   = tid >> 5;
  const int wm   = wv >> 1;        // 0..3 wave row
  const int wn   = wv & 1;         // 0..1 wave col
  const int l    = lane & 15;
  const int hp   = lane >> 4;      // half-wave K-block select

  // staging: 128x64 halfs per tile = 1024 b128 chunks; 4 chunks/thread/tile
  int srow[4], scol[4];
  const _Float16* gA[4];
  const _Float16* gB[4];
  unsigned lA[2][4], lB[2][4];
#pragma unroll
  for (int r = 0; r < 4; ++r) {
    const int c = tid + 256 * r;         // 0..1023
    srow[r] = c >> 3;                    // 0..127
    scol[r] = (c & 7) * 8;               // 0,8,...,56
    gA[r] = ah + (size_t)(im0 + srow[r]) * NN + scol[r];
    gB[r] = ah + (size_t)(in0 + srow[r]) * NN + scol[r];
#pragma unroll
    for (int b = 0; b < 2; ++b) {
      lA[b][r] = lds_off(&As[b][srow[r]][scol[r]]);
      lB[b][r] = lds_off(&Bs[b][srow[r]][scol[r]]);
    }
  }

  F32Acc acc[2][4];
#pragma unroll
  for (int mi = 0; mi < 2; ++mi)
#pragma unroll
    for (int ni = 0; ni < 4; ++ni)
#pragma unroll
      for (int r = 0; r < 8; ++r) acc[mi][ni].f[r] = 0.f;

  // ---- prologue: async-stage kk=0 into buffer 0 ----
#pragma unroll
  for (int r = 0; r < 4; ++r) {
    async_copy_b128(lA[0][r], gA[r]);
    async_copy_b128(lB[0][r], gB[r]);
  }
  wait_async0();
  __syncthreads();

  // ---- main loop, 2 buffers per trip so buffer index is compile-time ----
  for (int kk = 0; kk < NN; kk += 2 * KT) {
#pragma unroll
    for (int ph = 0; ph < 2; ++ph) {      // ph == current buffer (constant)
      const int kbase = kk + ph * KT;

      // async-stage next K-tile into the other buffer (overlaps WMMA)
      if (kbase + KT < NN) {
        const int kd = kbase + KT;
#pragma unroll
        for (int r = 0; r < 4; ++r) {
          async_copy_b128(lA[ph ^ 1][r], gA[r] + kd);
          async_copy_b128(lB[ph ^ 1][r], gB[r] + kd);
        }
      }

      // two K=32 sub-steps from current buffer
#pragma unroll
      for (int ks = 0; ks < KT; ks += 32) {
        F16Frag afr[2], bfr[4];
#pragma unroll
        for (int mi = 0; mi < 2; ++mi) {
          const int rr = wm * 32 + mi * 16 + l;
          afr[mi].v8[0] = *(const v8h*)&As[ph][rr][ks + hp * 8];
          afr[mi].v8[1] = *(const v8h*)&As[ph][rr][ks + hp * 8 + 16];
        }
#pragma unroll
        for (int ni = 0; ni < 4; ++ni) {
          const int rr = wn * 64 + ni * 16 + l;
          bfr[ni].v8[0] = *(const v8h*)&Bs[ph][rr][ks + hp * 8];
          bfr[ni].v8[1] = *(const v8h*)&Bs[ph][rr][ks + hp * 8 + 16];
        }
#pragma unroll
        for (int mi = 0; mi < 2; ++mi)
#pragma unroll
          for (int ni = 0; ni < 4; ++ni)
            acc[mi][ni].v = __builtin_amdgcn_wmma_f32_16x16x32_f16(
                false, afr[mi].v, false, bfr[ni].v,
                (short)0, acc[mi][ni].v, false, false);
      }

      // reads of buffer `ph` are ds-waited before the WMMAs; wait for next
      // tile's async data, one barrier, continue into the other buffer
      wait_async0();
      __syncthreads();
    }
  }

  // ---- epilogue: out[i, h*N + k] (reference transpose+reshape layout) ----
  const size_t ldo = (size_t)HH * NN;
#pragma unroll
  for (int mi = 0; mi < 2; ++mi) {
#pragma unroll
    for (int ni = 0; ni < 4; ++ni) {
      const int col = in0 + wn * 64 + ni * 16 + l;
      const size_t obase = (size_t)h * NN + col;
#pragma unroll
      for (int r = 0; r < 8; ++r) {
        const int rowi = im0 + wm * 32 + mi * 16 + hp * 8 + r;
        out[(size_t)rowi * ldo + obase] = acc[mi][ni].f[r];
      }
    }
  }
}

extern "C" void kernel_launch(void* const* d_in, const int* in_sizes, int n_in,
                              void* d_out, int out_size, void* d_ws, size_t ws_size,
                              hipStream_t stream) {
  const float* ft       = (const float*)d_in[0];  // (1, FT, N)
  const float* bias_mat = (const float*)d_in[1];  // (1, N, N)
  const float* W        = (const float*)d_in[2];  // (H, FT)
  const float* bvec     = (const float*)d_in[3];  // (H,)
  float* out = (float*)d_out;                     // (1, N, H*N) fp32

  // workspace layout: w (H*N f32, 64KB) | a (H*N*N f16, 128MB)
  float*    wbuf = (float*)d_ws;
  _Float16* abuf = (_Float16*)((char*)d_ws + 65536);

  gat_proj_kernel<<<dim3(NN / 256, HH), 256, 0, stream>>>(ft, W, bvec, wbuf);
  gat_softmax_kernel<<<dim3(HH * NN), 256, 0, stream>>>(bias_mat, wbuf, abuf);
  gat_syrk_kernel<<<dim3(NN / 128, NN / 128, HH), 256, 0, stream>>>(abuf, out);
}
